// DynamicGNNdec_30751965839950
// MI455X (gfx1250) — matrix operand order repaired and checked by reading, not compiled, not run on hardware
//
#include <hip/hip_runtime.h>
#include <hip/hip_bf16.h>

// ---------------------------------------------------------------------------
// Types for CDNA5 WMMA (wave32): bf16 A/B fragments, f32 accumulator
// ---------------------------------------------------------------------------
typedef __attribute__((ext_vector_type(16))) __bf16 v16bf;
typedef __attribute__((ext_vector_type(8)))  float  v8f;

union BF16Frag { v16bf v; unsigned int u[8]; };
union AccFrag  { v8f   v; float        f[8]; };

#ifndef __has_builtin
#define __has_builtin(x) 0
#endif
#if __has_builtin(__builtin_amdgcn_tensor_load_to_lds) && __has_builtin(__builtin_amdgcn_s_wait_tensorcnt)
#define USE_TDM 1
#else
#define USE_TDM 0
#endif

typedef __attribute__((ext_vector_type(4))) unsigned int v4u_;
typedef __attribute__((ext_vector_type(8))) int v8i_;
typedef __attribute__((ext_vector_type(4))) int v4i_;

__device__ __forceinline__ unsigned short f2bf(float x) {
  unsigned int u = __float_as_uint(x);
  unsigned int r = u + 0x7FFFu + ((u >> 16) & 1u);   // round-to-nearest-even
  return (unsigned short)(r >> 16);
}

#define BROW 130   // LDS row stride (uints) of a B tile row: 128 + 2 pad dwords

#if USE_TDM
// Issue a TDM 2D tile load: 16 pair-rows x 128 uint cols from a (Kpairs x Hout)
// uint matrix into LDS, with 2-dword padding every 512B (row stride = BROW).
// Hardware zero-fills reads beyond (remRows x remCols).
// NOTE: this toolchain's builtin arity is 6 (groups 0..3 + extra int32x8 + cpol).
__device__ __forceinline__ void tdm_load_b(const unsigned int* gsrc, unsigned int ldsAddr,
                                           int remCols, int remRows, int strideElems)
{
  unsigned long long ga = (unsigned long long)gsrc;
  v4u_ g0;
  g0[0] = 1u;                                           // count=1, user mode
  g0[1] = ldsAddr;                                      // lds_addr (bytes)
  g0[2] = (unsigned int)(ga & 0xFFFFFFFFu);             // global_addr[31:0]
  g0[3] = (unsigned int)((ga >> 32) & 0x01FFFFFFu)      // global_addr[56:32]
        | (2u << 30);                                   // type = 2 ("image")
  v8i_ g1;
  g1[0] = (int)((2u << 16)      // data_size = 4 bytes
              | (1u << 20)      // pad_enable
              | (6u << 22)      // pad_interval: 128 dwords (512B)
              | (1u << 25));    // pad_amount: 2 dwords
  g1[1] = (int)(((unsigned)remCols & 0xFFFFu) << 16);                      // tensor_dim0 lo
  g1[2] = (int)((((unsigned)remCols >> 16) & 0xFFFFu)
              | (((unsigned)remRows & 0xFFFFu) << 16));                    // dim0 hi | dim1 lo
  g1[3] = (int)((((unsigned)remRows >> 16) & 0xFFFFu) | (128u << 16));     // dim1 hi | tile_dim0=128
  g1[4] = (int)16u;                                                        // tile_dim1=16, tile_dim2=0
  g1[5] = (int)(unsigned)strideElems;                                      // tensor_dim0_stride lo
  g1[6] = 0;
  g1[7] = 0;
  v4i_ gz4 = {0, 0, 0, 0};
  v8i_ gz8 = {0, 0, 0, 0, 0, 0, 0, 0};
  __builtin_amdgcn_tensor_load_to_lds(g0, g1, gz4, gz4, gz8, 0);
}
#endif

// ---------------------------------------------------------------------------
// Generic WMMA GEMM: out[M x Hout] = act( A[M x Kdim] @ W + bias )
// W is pre-packed as k-pairs: Wpk[(k/2)*Hout + c] = {bf16 W[k,c], bf16 W[k+1,c]}
// knn != nullptr: gather mode (EdgeConv messages):
//   A[gm, k] = (k < F) ? x[node, k] : x[nbr, k-F] - x[node, k-F]
// Block tile 64 rows x 128 cols; 8 waves = 4 row-tiles x 2 col-halves;
// each wave: 4 accumulators -> 4 v_wmma_f32_16x16x32_bf16 per 32-wide k step.
// B chunks are double-buffered in LDS via the Tensor Data Mover (async),
// synchronized with s_wait_tensorcnt; fallback: cooperative staging.
// ---------------------------------------------------------------------------
__global__ void __launch_bounds__(256)
gemm_bf16_wmma(const float* __restrict__ xin, const int* __restrict__ knn,
               const unsigned int* __restrict__ Wpk, const float* __restrict__ bias,
               float* __restrict__ out,
               int Mtot, int n_per_batch, int F, int Kdim, int Hout, int KNN, int act)
{
  __shared__ unsigned short Alds[64 * 40];        // 64 rows x 32 k, stride 40
  __shared__ unsigned int   Bt[2][16 * BROW];     // double-buffered B pair tiles

  const int tid  = threadIdx.x;
  const int lane = tid & 31;
  const int wave = tid >> 5;
  const int rt   = wave & 3;       // row tile 0..3
  const int ch   = wave >> 2;      // col half 0..1 (4 x 16-col tiles each)
  const int m0   = blockIdx.x * 64;
  const int c0   = blockIdx.y * 128;

  v8f acc0 = {}, acc1 = {}, acc2 = {}, acc3 = {};

  const int nk = (Kdim + 31) >> 5;

  // ---- A staging thread mapping: thread -> (row, 8 consecutive k) ----
  const int ar = tid >> 2;             // 0..63
  const int ak = (tid & 3) << 3;       // 0,8,16,24
  const int gm = m0 + ar;
  const bool rowValid = (gm < Mtot);
  int nodeRow = 0, nbrRow = 0;
  if (knn && rowValid) {
    int node = gm / KNN;
    int jj   = gm - node * KNN;
    int base = (node / n_per_batch) * n_per_batch;
    nodeRow = node;
    nbrRow  = base + knn[node * KNN + jj];
  }
  const bool vecOK = ((F & 3) == 0);

#if USE_TDM
  const int Kpairs = Kdim >> 1;
  if (wave == 0) {
    tdm_load_b(Wpk + (size_t)0 * Hout + c0,
               (unsigned int)(unsigned long long)&Bt[0][0],
               Hout - c0, Kpairs, Hout);
  }
#endif

  for (int kc = 0; kc < nk; ++kc) {
    const int kbase = kc << 5;
    const int buf   = kc & 1;

    // ---- stage A (f32 -> bf16), specialized to avoid divergent selects ----
    {
      const int kg0 = kbase + ak;
      unsigned short* dst = &Alds[ar * 40 + ak];
      if (!rowValid) {
        #pragma unroll
        for (int e = 0; e < 8; ++e) dst[e] = 0;
      } else if (knn == nullptr) {
        if ((Kdim & 3) == 0 && kg0 + 8 <= Kdim) {
          const float4* p = (const float4*)(xin + (size_t)gm * Kdim + kg0);
          float4 a = p[0], b4 = p[1];
          dst[0] = f2bf(a.x);  dst[1] = f2bf(a.y);  dst[2] = f2bf(a.z);  dst[3] = f2bf(a.w);
          dst[4] = f2bf(b4.x); dst[5] = f2bf(b4.y); dst[6] = f2bf(b4.z); dst[7] = f2bf(b4.w);
        } else {
          #pragma unroll
          for (int e = 0; e < 8; ++e) {
            int kg = kg0 + e;
            dst[e] = (kg < Kdim) ? f2bf(xin[(size_t)gm * Kdim + kg]) : (unsigned short)0;
          }
        }
      } else if (kg0 + 8 <= F) {                       // pure x_i half
        if (vecOK) {
          const float4* p = (const float4*)(xin + (size_t)nodeRow * F + kg0);
          float4 a = p[0], b4 = p[1];
          dst[0] = f2bf(a.x);  dst[1] = f2bf(a.y);  dst[2] = f2bf(a.z);  dst[3] = f2bf(a.w);
          dst[4] = f2bf(b4.x); dst[5] = f2bf(b4.y); dst[6] = f2bf(b4.z); dst[7] = f2bf(b4.w);
        } else {
          #pragma unroll
          for (int e = 0; e < 8; ++e) dst[e] = f2bf(xin[(size_t)nodeRow * F + kg0 + e]);
        }
      } else if (kg0 >= F) {                           // pure (x_j - x_i) half
        if (vecOK && kg0 + 8 <= Kdim) {
          const float4* pj = (const float4*)(xin + (size_t)nbrRow  * F + (kg0 - F));
          const float4* pi = (const float4*)(xin + (size_t)nodeRow * F + (kg0 - F));
          float4 aj = pj[0], bj = pj[1], ai = pi[0], bi2 = pi[1];
          dst[0] = f2bf(aj.x - ai.x);  dst[1] = f2bf(aj.y - ai.y);
          dst[2] = f2bf(aj.z - ai.z);  dst[3] = f2bf(aj.w - ai.w);
          dst[4] = f2bf(bj.x - bi2.x); dst[5] = f2bf(bj.y - bi2.y);
          dst[6] = f2bf(bj.z - bi2.z); dst[7] = f2bf(bj.w - bi2.w);
        } else {
          #pragma unroll
          for (int e = 0; e < 8; ++e) {
            int kg = kg0 + e;
            dst[e] = (kg < Kdim)
                   ? f2bf(xin[(size_t)nbrRow * F + (kg - F)] - xin[(size_t)nodeRow * F + (kg - F)])
                   : (unsigned short)0;
          }
        }
      } else {                                         // chunk straddles F (only Kdim=78 path)
        #pragma unroll
        for (int e = 0; e < 8; ++e) {
          int kg = kg0 + e;
          float val = 0.f;
          if (kg < Kdim) {
            if (kg < F) val = xin[(size_t)nodeRow * F + kg];
            else { int f = kg - F; val = xin[(size_t)nbrRow * F + f] - xin[(size_t)nodeRow * F + f]; }
          }
          dst[e] = f2bf(val);
        }
      }
    }

    // ---- B tile: TDM double-buffer (async) or cooperative fallback ----
#if USE_TDM
    if (wave == 0) {
      if (kc + 1 < nk) {
        tdm_load_b(Wpk + (size_t)((kbase + 32) >> 1) * Hout + c0,
                   (unsigned int)(unsigned long long)&Bt[buf ^ 1][0],
                   Hout - c0, Kpairs - ((kbase + 32) >> 1), Hout);
        __builtin_amdgcn_s_wait_tensorcnt(1);   // chunk kc resident, kc+1 in flight
      } else {
        __builtin_amdgcn_s_wait_tensorcnt(0);
      }
    }
    asm volatile("" ::: "memory");
#else
    {
      const int Kpairs = Kdim >> 1;
      int pr = tid >> 4;                   // pair-row 0..15
      int cb = (tid & 15) << 3;            // col 0..120
      int kkg = (kbase >> 1) + pr;
      #pragma unroll
      for (int cc = 0; cc < 8; ++cc) {
        int col = c0 + cb + cc;
        unsigned int vv = 0;
        if (kkg < Kpairs && col < Hout) vv = Wpk[(size_t)kkg * Hout + col];
        Bt[buf][pr * BROW + cb + cc] = vv;
      }
    }
#endif
    __syncthreads();

    // ---- fragments per ISA 16-bit A(16x32) / B(32x16) layouts ----
    BF16Frag af;
    {
      const int m = (rt << 4) + (lane & 15);
      #pragma unroll
      for (int v = 0; v < 8; ++v) {
        int k0 = ((v & 3) << 1) + ((v & 4) ? 16 : 0) + ((lane & 16) ? 8 : 0);
        af.u[v] = *(const unsigned int*)&Alds[m * 40 + k0];
      }
    }
    {
      const unsigned int* Bb = &Bt[buf][0];
      const int kkoff = (lane & 16) ? 4 : 0;
      const int cbase = (ch << 6) + (lane & 15);
      BF16Frag bf;
      #pragma unroll
      for (int v = 0; v < 8; ++v) bf.u[v] = Bb[(kkoff + (v & 3) + ((v & 4) << 1)) * BROW + cbase];
      acc0 = __builtin_amdgcn_wmma_f32_16x16x32_bf16(false, af.v, false, bf.v, (short)0, acc0, false, false);
      #pragma unroll
      for (int v = 0; v < 8; ++v) bf.u[v] = Bb[(kkoff + (v & 3) + ((v & 4) << 1)) * BROW + cbase + 16];
      acc1 = __builtin_amdgcn_wmma_f32_16x16x32_bf16(false, af.v, false, bf.v, (short)0, acc1, false, false);
      #pragma unroll
      for (int v = 0; v < 8; ++v) bf.u[v] = Bb[(kkoff + (v & 3) + ((v & 4) << 1)) * BROW + cbase + 32];
      acc2 = __builtin_amdgcn_wmma_f32_16x16x32_bf16(false, af.v, false, bf.v, (short)0, acc2, false, false);
      #pragma unroll
      for (int v = 0; v < 8; ++v) bf.u[v] = Bb[(kkoff + (v & 3) + ((v & 4) << 1)) * BROW + cbase + 48];
      acc3 = __builtin_amdgcn_wmma_f32_16x16x32_bf16(false, af.v, false, bf.v, (short)0, acc3, false, false);
    }
    __syncthreads();
  }

  // ---- store per C/D layout: VGPR i -> M = i (+8 for upper half-wave) ----
  const int mbase = m0 + (rt << 4) + ((lane & 16) ? 8 : 0);
  v8f accs[4] = {acc0, acc1, acc2, acc3};
  #pragma unroll
  for (int t = 0; t < 4; ++t) {
    AccFrag R; R.v = accs[t];
    int col = c0 + (ch << 6) + (t << 4) + (lane & 15);
    if (col < Hout) {
      float bcol = bias[col];
      #pragma unroll
      for (int i = 0; i < 8; ++i) {
        int row = mbase + i;
        if (row < Mtot) {
          float vv = R.f[i] + bcol;
          if (act == 1) vv = fmaxf(vv, 0.f);
          out[(size_t)row * Hout + col] = vv;
        }
      }
    }
  }
}

// ---------------------------------------------------------------------------
// kNN: per-batch pairwise squared distances (LDS-staged) + K smallest per row
// ---------------------------------------------------------------------------
__global__ void __launch_bounds__(256)
knn_kernel(const float* __restrict__ x, int* __restrict__ knn, int n, int F, int KNN)
{
  __shared__ float xs[39 * 256];
  __shared__ float ds[39 * 40];
  __shared__ float sq[40];
  const int b = blockIdx.x;
  const int tid = threadIdx.x;
  const int base = b * n;

  for (int t = tid; t < n * F; t += 256) xs[t] = x[base * F + t];
  __syncthreads();
  for (int i = tid; i < n; i += 256) {
    float s = 0.f;
    for (int f = 0; f < F; ++f) { float a = xs[i * F + f]; s += a * a; }
    sq[i] = s;
  }
  __syncthreads();
  for (int p = tid; p < n * n; p += 256) {
    int i = p / n, j = p - i * n;
    float dot = 0.f;
    for (int f = 0; f < F; ++f) dot += xs[i * F + f] * xs[j * F + f];
    ds[i * 40 + j] = sq[i] + sq[j] - 2.f * dot;
  }
  __syncthreads();
  for (int i = tid; i < n; i += 256) {
    for (int q = 0; q < KNN; ++q) {
      float best = 3.4e38f; int bj = 0;
      for (int j = 0; j < n; ++j) {
        float d = ds[i * 40 + j];
        if (d < best) { best = d; bj = j; }
      }
      ds[i * 40 + bj] = 3.4e38f;
      knn[(base + i) * KNN + q] = bj;
    }
  }
}

// max over K neighbor messages of relu(msg); optional tanh (decoder layers)
__global__ void edge_reduce(const float* __restrict__ msg, float* __restrict__ out,
                            int nodes, int Hout, int KNN, int mode)
{
  int t = blockIdx.x * blockDim.x + threadIdx.x;
  if (t >= nodes * Hout) return;
  int node = t / Hout, h = t - node * Hout;
  float m = 0.f;
  for (int j = 0; j < KNN; ++j)
    m = fmaxf(m, msg[(size_t)(node * KNN + j) * Hout + h]);
  if (mode == 1) m = tanhf(m);
  out[t] = m;
}

// batchnorm over rows (population var) + relu, in place; one block per column
__global__ void __launch_bounds__(256)
bn_relu_kernel(float* __restrict__ x, const float* __restrict__ g,
               const float* __restrict__ bb, int M, int H)
{
  __shared__ float s1[256], s2[256];
  int c = blockIdx.x, tid = threadIdx.x;
  float a = 0.f, b2 = 0.f;
  for (int r = tid; r < M; r += 256) { float v = x[r * H + c]; a += v; b2 += v * v; }
  s1[tid] = a; s2[tid] = b2;
  __syncthreads();
  for (int s = 128; s > 0; s >>= 1) {
    if (tid < s) { s1[tid] += s1[tid + s]; s2[tid] += s2[tid + s]; }
    __syncthreads();
  }
  float mean = s1[0] / M;
  float var  = s2[0] / M - mean * mean;
  float rstd = rsqrtf(var + 1e-5f);
  float gg = g[c], bc = bb[c];
  for (int r = tid; r < M; r += 256) {
    float v = (x[r * H + c] - mean) * rstd * gg + bc;
    x[r * H + c] = fmaxf(v, 0.f);
  }
}

// GCN score: s = D^-1/2 (A+I) D^-1/2 (x @ w) + b ; one block per batch
__global__ void __launch_bounds__(64)
gcn_score_kernel(const float* __restrict__ x, const float* __restrict__ adj,
                 const float* __restrict__ w, const float* __restrict__ b,
                 float* __restrict__ score, int n, int F)
{
  __shared__ float s0[40], dinv[40];
  int bi = blockIdx.x, i = threadIdx.x;
  const float* A = adj + bi * n * n;
  if (i < n) {
    float s = 0.f;
    for (int f = 0; f < F; ++f) s += x[(bi * n + i) * F + f] * w[f];
    s0[i] = s;
    float rs = 1.f;
    for (int j = 0; j < n; ++j) rs += A[i * n + j];
    dinv[i] = rsqrtf(rs);
  }
  __syncthreads();
  if (i < n) {
    float acc = dinv[i] * s0[i];
    for (int j = 0; j < n; ++j) acc += A[i * n + j] * dinv[j] * s0[j];
    score[bi * n + i] = dinv[i] * acc + b[0];
  }
}

// top-k descending (ties -> lower index, matching jax top_k)
__global__ void topk_kernel(const float* __restrict__ score, int n, int k,
                            int* __restrict__ loc, float* __restrict__ topv,
                            int* __restrict__ perm, int nb)
{
  int b = blockIdx.x * blockDim.x + threadIdx.x;
  if (b >= nb) return;
  float s[40];
  for (int i = 0; i < n; ++i) s[i] = score[b * n + i];
  for (int q = 0; q < k; ++q) {
    float best = -3.4e38f; int bi2 = 0;
    for (int i = 0; i < n; ++i) if (s[i] > best) { best = s[i]; bi2 = i; }
    s[bi2] = -3.4e38f;
    loc[b * k + q]  = bi2;
    topv[b * k + q] = best;
    perm[b * k + q] = b * n + bi2;
  }
}

__global__ void sag_apply_kernel(const float* __restrict__ x, const int* __restrict__ loc,
                                 const float* __restrict__ topv, float* __restrict__ xp,
                                 int nb, int n, int k, int H)
{
  int t = blockIdx.x * blockDim.x + threadIdx.x;
  if (t >= nb * k * H) return;
  int h = t % H, row = t / H, b = row / k;
  xp[row * H + h] = x[(b * n + loc[row]) * H + h] * tanhf(topv[row]);
}

__global__ void adj_gather_kernel(const float* __restrict__ adj, const int* __restrict__ loc,
                                  float* __restrict__ adjp, int nb, int n, int k)
{
  int t = blockIdx.x * blockDim.x + threadIdx.x;
  if (t >= nb * k * k) return;
  int q = t % k, r = (t / k) % k, b = t / (k * k);
  adjp[t] = adj[b * n * n + loc[b * k + r] * n + loc[b * k + q]];
}

__global__ void zero_kernel(float* p, int count) {
  int t = blockIdx.x * blockDim.x + threadIdx.x;
  if (t < count) p[t] = 0.f;
}

__global__ void scatter_rows_kernel(float* __restrict__ dst, const float* __restrict__ src,
                                    const int* __restrict__ perm, int rows, int H)
{
  int t = blockIdx.x * blockDim.x + threadIdx.x;
  if (t >= rows * H) return;
  int h = t % H, r = t / H;
  dst[perm[r] * H + h] = src[t];
}

__global__ void readout_kernel(const float* __restrict__ xp, float* __restrict__ z,
                               int k, int H, int mode)
{
  int b = blockIdx.x, h = threadIdx.x;
  if (h >= H) return;
  float mx = -3.4e38f, sm = 0.f;
  for (int q = 0; q < k; ++q) {
    float v = xp[(b * k + q) * H + h];
    mx = fmaxf(mx, v); sm += v;
  }
  float mean = sm / k;
  if (mode == 0) { z[b * 2 * H + h] = mx; z[b * 2 * H + H + h] = mean; }
  else           { z[b * 2 * H + h] += mx; z[b * 2 * H + H + h] += mean; }
}

__global__ void deg_kernel(const float* __restrict__ adj, float* __restrict__ deg,
                           int nb, int n)
{
  int t = blockIdx.x * blockDim.x + threadIdx.x;
  if (t >= nb * n) return;
  float s = 0.f;
  for (int j = 0; j < n; ++j) s += adj[t * n + j];
  deg[t] = s;
}

__global__ void vecdot_relu_kernel(const float* __restrict__ x, const float* __restrict__ w,
                                   const float* __restrict__ b, float* __restrict__ out,
                                   int M, int K)
{
  int t = blockIdx.x * blockDim.x + threadIdx.x;
  if (t >= M) return;
  float s = b[0];
  for (int f = 0; f < K; ++f) s += x[t * K + f] * w[f];
  out[t] = fmaxf(s, 0.f);
}

__global__ void gather_gtpred_kernel(const int* __restrict__ perm, const float* __restrict__ deg,
                                     const float* __restrict__ pred, float* __restrict__ gt_out,
                                     float* __restrict__ pred_out, int len)
{
  int t = blockIdx.x * blockDim.x + threadIdx.x;
  if (t >= len) return;
  int p = perm[t];
  gt_out[t] = deg[p];
  pred_out[t] = pred[p];
}

__global__ void lin3_softmax_kernel(const float* __restrict__ x, const float* __restrict__ w,
                                    const float* __restrict__ b, float* __restrict__ out,
                                    int M, int K)
{
  int t = blockIdx.x * blockDim.x + threadIdx.x;
  if (t >= M) return;
  float l0 = b[0], l1 = b[1];
  for (int f = 0; f < K; ++f) {
    float v = x[t * K + f];
    l0 += v * w[f * 2 + 0];
    l1 += v * w[f * 2 + 1];
  }
  float m = fmaxf(l0, l1);
  float e0 = expf(l0 - m), e1 = expf(l1 - m);
  float s = e0 + e1;
  out[t * 2 + 0] = e0 / s;
  out[t * 2 + 1] = e1 / s;
}

// pack f32 weights (Kdim x Hout, Kdim even) into k-pair bf16 uints:
// out[(k/2)*Hout + c] = bf16(W[k,c]) | bf16(W[k+1,c]) << 16
__global__ void cvt_pack_kernel(const float* __restrict__ in, unsigned int* __restrict__ out,
                                int Kpairs, int Hout)
{
  int t = blockIdx.x * blockDim.x + threadIdx.x;
  if (t >= Kpairs * Hout) return;
  int kk = t / Hout, c = t - kk * Hout;
  unsigned int lo = f2bf(in[(size_t)(2 * kk) * Hout + c]);
  unsigned int hi = f2bf(in[(size_t)(2 * kk + 1) * Hout + c]);
  out[t] = lo | (hi << 16);
}

// ---------------------------------------------------------------------------
// Host orchestration
// ---------------------------------------------------------------------------
extern "C" void kernel_launch(void* const* d_in, const int* in_sizes, int n_in,
                              void* d_out, int out_size, void* d_ws, size_t ws_size,
                              hipStream_t stream)
{
  (void)in_sizes; (void)n_in; (void)out_size; (void)ws_size;
  const int B_ = 128, N_ = 39, KNN_ = 10, H_ = 256;
  const int K1_ = 32, K2_ = 26, K3_ = 21;
  const int BN = B_ * N_;

  const float* x_in   = (const float*)d_in[0];
  const float* adj_in = (const float*)d_in[1];
  const float* w_mlp1 = (const float*)d_in[2];  const float* b_mlp1 = (const float*)d_in[3];
  const float* w_mlp2 = (const float*)d_in[4];  const float* b_mlp2 = (const float*)d_in[5];
  const float* w_mlp3 = (const float*)d_in[6];  const float* b_mlp3 = (const float*)d_in[7];
  const float* w_mlp6 = (const float*)d_in[8];  const float* b_mlp6 = (const float*)d_in[9];
  const float* bn1_g = (const float*)d_in[10];  const float* bn1_b = (const float*)d_in[11];
  const float* bn2_g = (const float*)d_in[12];  const float* bn2_b = (const float*)d_in[13];
  const float* bn3_g = (const float*)d_in[14];  const float* bn3_b = (const float*)d_in[15];
  const float* p1_w = (const float*)d_in[16];   const float* p1_b = (const float*)d_in[17];
  const float* p2_w = (const float*)d_in[18];   const float* p2_b = (const float*)d_in[19];
  const float* p3_w = (const float*)d_in[20];   const float* p3_b = (const float*)d_in[21];
  const float* lin1_w = (const float*)d_in[22]; const float* lin1_b = (const float*)d_in[23];
  const float* lin2_w = (const float*)d_in[24]; const float* lin2_b = (const float*)d_in[25];
  const float* lin3_w = (const float*)d_in[26]; const float* lin3_b = (const float*)d_in[27];
  const float* lin4_w = (const float*)d_in[28]; const float* lin4_b = (const float*)d_in[29];
  const float* lin5_w = (const float*)d_in[30]; const float* lin5_b = (const float*)d_in[31];
  const float* lin6_w = (const float*)d_in[32]; const float* lin6_b = (const float*)d_in[33];

  float* out = (float*)d_out;
  float* o_probs = out;
  float* o_xdec1 = out + 256;
  float* o_xdec2 = o_xdec1 + BN * N_;
  float* o_xdec3 = o_xdec2 + BN * N_;
  float* o_gt1   = o_xdec3 + BN * N_;
  float* o_pred1 = o_gt1 + B_ * K1_;
  float* o_gt2   = o_pred1 + B_ * K1_;
  float* o_pred2 = o_gt2 + B_ * K2_;
  float* o_gt3   = o_pred2 + B_ * K2_;
  float* o_pred3 = o_gt3 + B_ * K3_;

  char* ws = (char*)d_ws;
  size_t off = 0;
  auto alloc = [&](size_t bytes) -> char* {
    char* p = ws + off;
    off += (bytes + 255) & ~(size_t)255;
    return p;
  };
  // packed k-pair weights (uint per pair, size == Kdim*Hout*2 bytes)
  unsigned int* wb1  = (unsigned int*)alloc((size_t)39  * 256 * 4);
  unsigned int* wb2  = (unsigned int*)alloc((size_t)256 * 256 * 4);
  unsigned int* wb3  = (unsigned int*)alloc((size_t)256 * 256 * 4);
  unsigned int* wb6  = (unsigned int*)alloc((size_t)256 * 39  * 4);
  unsigned int* wbl1 = (unsigned int*)alloc((size_t)256 * 256 * 4);
  unsigned int* wbl2 = (unsigned int*)alloc((size_t)128 * 128 * 4);
  unsigned int* wbl4 = (unsigned int*)alloc((size_t)128 * 256 * 4);
  unsigned int* wbl5 = (unsigned int*)alloc((size_t)128 * 128 * 4);
  float* msgbuf = (float*)alloc((size_t)BN * KNN_ * H_ * 4);
  int*   knnbuf = (int*)alloc((size_t)BN * KNN_ * 4);
  float* h1  = (float*)alloc((size_t)BN * H_ * 4);
  float* h2  = (float*)alloc((size_t)B_ * K1_ * H_ * 4);
  float* h3  = (float*)alloc((size_t)B_ * K2_ * H_ * 4);
  float* xp1 = (float*)alloc((size_t)B_ * K1_ * H_ * 4);
  float* xp2 = (float*)alloc((size_t)B_ * K2_ * H_ * 4);
  float* xp3 = (float*)alloc((size_t)B_ * K3_ * H_ * 4);
  float* xout  = (float*)alloc((size_t)BN * H_ * 4);
  float* dbuf1 = (float*)alloc((size_t)BN * H_ * 4);
  float* dbuf2 = (float*)alloc((size_t)BN * H_ * 4);
  float* dh_a  = (float*)alloc((size_t)BN * H_ * 4);
  float* dh_b  = (float*)alloc((size_t)BN * 128 * 4);
  float* x_degree = (float*)alloc((size_t)BN * 4);
  float* scores = (float*)alloc((size_t)BN * 4);
  int*   locbuf = (int*)alloc((size_t)B_ * K1_ * 4);
  float* topv   = (float*)alloc((size_t)B_ * K1_ * 4);
  int*   perm1  = (int*)alloc((size_t)B_ * K1_ * 4);
  int*   perm2  = (int*)alloc((size_t)B_ * K2_ * 4);
  int*   perm3  = (int*)alloc((size_t)B_ * K3_ * 4);
  float* adj1 = (float*)alloc((size_t)B_ * K1_ * K1_ * 4);
  float* adj2 = (float*)alloc((size_t)B_ * K2_ * K2_ * 4);
  float* deg_gt = (float*)alloc((size_t)BN * 4);
  float* z  = (float*)alloc((size_t)B_ * 512 * 4);
  float* z1 = (float*)alloc((size_t)B_ * 256 * 4);
  float* z2 = (float*)alloc((size_t)B_ * 128 * 4);

  auto gemm = [&](const float* X, const int* knn, const unsigned int* Wb,
                  const float* bias, float* o, int Mtot, int npb, int F, int Kd,
                  int Ho, int knnK, int act) {
    dim3 g((Mtot + 63) / 64, (Ho + 127) / 128);
    gemm_bf16_wmma<<<g, 256, 0, stream>>>(X, knn, Wb, bias, o, Mtot, npb, F, Kd, Ho, knnK, act);
  };
  auto zero = [&](float* p, int cnt) {
    zero_kernel<<<(cnt + 255) / 256, 256, 0, stream>>>(p, cnt);
  };
  auto scatter = [&](float* dst, const float* src, const int* perm, int rows, int Hc) {
    scatter_rows_kernel<<<(rows * Hc + 255) / 256, 256, 0, stream>>>(dst, src, perm, rows, Hc);
  };
  auto decoder = [&](const float* xf, float* xdec) {
    knn_kernel<<<B_, 256, 0, stream>>>(xf, knnbuf, N_, H_, KNN_);
    gemm(xf, knnbuf, wb3, b_mlp3, msgbuf, BN * KNN_, N_, H_, 2 * H_, H_, KNN_, 0);
    edge_reduce<<<(BN * H_ + 255) / 256, 256, 0, stream>>>(msgbuf, dbuf1, BN, H_, KNN_, 1);
    knn_kernel<<<B_, 256, 0, stream>>>(dbuf1, knnbuf, N_, H_, KNN_);
    gemm(dbuf1, knnbuf, wb2, b_mlp2, msgbuf, BN * KNN_, N_, H_, 2 * H_, H_, KNN_, 0);
    edge_reduce<<<(BN * H_ + 255) / 256, 256, 0, stream>>>(msgbuf, dbuf2, BN, H_, KNN_, 1);
    knn_kernel<<<B_, 256, 0, stream>>>(dbuf2, knnbuf, N_, H_, KNN_);
    gemm(dbuf2, knnbuf, wb6, b_mlp6, msgbuf, BN * KNN_, N_, H_, 2 * H_, N_, KNN_, 0);
    edge_reduce<<<(BN * N_ + 255) / 256, 256, 0, stream>>>(msgbuf, xdec, BN, N_, KNN_, 0);
  };

  auto cvt = [&](const float* w, unsigned int* dst, int Kpairs, int Ho) {
    cvt_pack_kernel<<<(Kpairs * Ho + 255) / 256, 256, 0, stream>>>(w, dst, Kpairs, Ho);
  };
  cvt(w_mlp1, wb1, 39, 256);   cvt(w_mlp2, wb2, 256, 256);
  cvt(w_mlp3, wb3, 256, 256);  cvt(w_mlp6, wb6, 256, 39);
  cvt(lin1_w, wbl1, 256, 256); cvt(lin2_w, wbl2, 128, 128);
  cvt(lin4_w, wbl4, 128, 256); cvt(lin5_w, wbl5, 128, 128);

  deg_kernel<<<(BN + 255) / 256, 256, 0, stream>>>(adj_in, deg_gt, B_, N_);

  // ================= Level 1 =================
  knn_kernel<<<B_, 256, 0, stream>>>(x_in, knnbuf, N_, N_, KNN_);
  gemm(x_in, knnbuf, wb1, b_mlp1, msgbuf, BN * KNN_, N_, N_, 2 * N_, H_, KNN_, 0);
  edge_reduce<<<(BN * H_ + 255) / 256, 256, 0, stream>>>(msgbuf, h1, BN, H_, KNN_, 0);
  bn_relu_kernel<<<H_, 256, 0, stream>>>(h1, bn1_g, bn1_b, BN, H_);
  gcn_score_kernel<<<B_, 64, 0, stream>>>(h1, adj_in, p1_w, p1_b, scores, N_, H_);
  topk_kernel<<<(B_ + 63) / 64, 64, 0, stream>>>(scores, N_, K1_, locbuf, topv, perm1, B_);
  sag_apply_kernel<<<(B_ * K1_ * H_ + 255) / 256, 256, 0, stream>>>(h1, locbuf, topv, xp1, B_, N_, K1_, H_);
  adj_gather_kernel<<<(B_ * K1_ * K1_ + 255) / 256, 256, 0, stream>>>(adj_in, locbuf, adj1, B_, N_, K1_);
  zero(xout, BN * H_);
  scatter(xout, xp1, perm1, B_ * K1_, H_);
  decoder(xout, o_xdec1);
  gemm(xout, nullptr, wbl4, lin4_b, dh_a, BN, 1, 0, 256, 256, 1, 1);
  gemm(dh_a, nullptr, wbl5, lin5_b, dh_b, BN, 1, 0, 256, 128, 1, 1);
  vecdot_relu_kernel<<<(BN + 255) / 256, 256, 0, stream>>>(dh_b, lin6_w, lin6_b, x_degree, BN, 128);
  gather_gtpred_kernel<<<(B_ * K1_ + 255) / 256, 256, 0, stream>>>(perm1, deg_gt, x_degree, o_gt1, o_pred1, B_ * K1_);
  readout_kernel<<<B_, 256, 0, stream>>>(xp1, z, K1_, H_, 0);

  // ================= Level 2 =================
  knn_kernel<<<B_, 256, 0, stream>>>(xp1, knnbuf, K1_, H_, KNN_);
  gemm(xp1, knnbuf, wb2, b_mlp2, msgbuf, B_ * K1_ * KNN_, K1_, H_, 2 * H_, H_, KNN_, 0);
  edge_reduce<<<(B_ * K1_ * H_ + 255) / 256, 256, 0, stream>>>(msgbuf, h2, B_ * K1_, H_, KNN_, 0);
  bn_relu_kernel<<<H_, 256, 0, stream>>>(h2, bn2_g, bn2_b, B_ * K1_, H_);
  gcn_score_kernel<<<B_, 64, 0, stream>>>(h2, adj1, p2_w, p2_b, scores, K1_, H_);
  topk_kernel<<<(B_ + 63) / 64, 64, 0, stream>>>(scores, K1_, K2_, locbuf, topv, perm2, B_);
  sag_apply_kernel<<<(B_ * K2_ * H_ + 255) / 256, 256, 0, stream>>>(h2, locbuf, topv, xp2, B_, K1_, K2_, H_);
  adj_gather_kernel<<<(B_ * K2_ * K2_ + 255) / 256, 256, 0, stream>>>(adj1, locbuf, adj2, B_, K1_, K2_);
  zero(dbuf1, B_ * K1_ * H_);
  scatter(dbuf1, xp2, perm2, B_ * K2_, H_);
  zero(xout, BN * H_);
  scatter(xout, dbuf1, perm1, B_ * K1_, H_);
  decoder(xout, o_xdec2);
  gather_gtpred_kernel<<<(B_ * K2_ + 255) / 256, 256, 0, stream>>>(perm2, deg_gt, x_degree, o_gt2, o_pred2, B_ * K2_);
  readout_kernel<<<B_, 256, 0, stream>>>(xp2, z, K2_, H_, 1);

  // ================= Level 3 =================
  knn_kernel<<<B_, 256, 0, stream>>>(xp2, knnbuf, K2_, H_, KNN_);
  gemm(xp2, knnbuf, wb3, b_mlp3, msgbuf, B_ * K2_ * KNN_, K2_, H_, 2 * H_, H_, KNN_, 0);
  edge_reduce<<<(B_ * K2_ * H_ + 255) / 256, 256, 0, stream>>>(msgbuf, h3, B_ * K2_, H_, KNN_, 0);
  bn_relu_kernel<<<H_, 256, 0, stream>>>(h3, bn3_g, bn3_b, B_ * K2_, H_);
  gcn_score_kernel<<<B_, 64, 0, stream>>>(h3, adj2, p3_w, p3_b, scores, K2_, H_);
  topk_kernel<<<(B_ + 63) / 64, 64, 0, stream>>>(scores, K2_, K3_, locbuf, topv, perm3, B_);
  sag_apply_kernel<<<(B_ * K3_ * H_ + 255) / 256, 256, 0, stream>>>(h3, locbuf, topv, xp3, B_, K2_, K3_, H_);
  zero(dbuf1, B_ * K2_ * H_);
  scatter(dbuf1, xp3, perm3, B_ * K3_, H_);
  zero(xout, BN * H_);
  scatter(xout, dbuf1, perm2, B_ * K2_, H_);
  decoder(xout, o_xdec3);
  gather_gtpred_kernel<<<(B_ * K3_ + 255) / 256, 256, 0, stream>>>(perm3, deg_gt, x_degree, o_gt3, o_pred3, B_ * K3_);
  readout_kernel<<<B_, 256, 0, stream>>>(xp3, z, K3_, H_, 1);

  // ================= Classifier =================
  gemm(z,  nullptr, wbl1, lin1_b, z1, B_, 1, 0, 512, 256, 1, 1);
  gemm(z1, nullptr, wbl2, lin2_b, z2, B_, 1, 0, 256, 128, 1, 1);
  lin3_softmax_kernel<<<(B_ + 63) / 64, 64, 0, stream>>>(z2, lin3_w, lin3_b, o_probs, B_, 128);
}
